// BatchelorFwd_64458869179022
// MI455X (gfx1250) — matrix operand order repaired and checked by reading, not compiled, not run on hardware
//
#include <hip/hip_runtime.h>
#include <stdint.h>

// Problem dims (fixed by reference)
#define NXD 320
#define NYD 320
#define NCD 12
#define NTD 16
#define NPIX (NXD * NYD)   // 102400
#define LDS_ROW 40         // halves per staged B row (80 B, 16B-aligned, padded)

typedef __attribute__((ext_vector_type(16))) _Float16 v16h;
typedef __attribute__((ext_vector_type(8)))  float    v8f;

// ---------------------------------------------------------------------------
// CDNA5 async global->LDS path (ASYNCcnt-tracked), guarded so the build is
// green on toolchains without the builtins.
#if defined(__has_builtin)
#  if __has_builtin(__builtin_amdgcn_global_load_async_to_lds_b128)
#    define HAVE_ASYNC_LDS 1
#  endif
#endif
#ifndef HAVE_ASYNC_LDS
#  define HAVE_ASYNC_LDS 0
#endif

static __device__ __forceinline__ void copy16B_to_lds(const _Float16* g, _Float16* l) {
#if HAVE_ASYNC_LDS
  typedef __attribute__((ext_vector_type(4))) int i4;
  typedef __attribute__((address_space(1))) i4 gi4;   // int4 in global AS
  typedef __attribute__((address_space(3))) i4 li4;   // int4 in LDS AS
  __builtin_amdgcn_global_load_async_to_lds_b128((gi4*)g, (li4*)l, 0, 0);
#else
  *(uint4*)l = *(const uint4*)g;   // global_load_b128 + ds_store_b128
#endif
}

static __device__ __forceinline__ void wait_async_lds() {
#if HAVE_ASYNC_LDS
#  if defined(__has_builtin) && __has_builtin(__builtin_amdgcn_s_wait_asynccnt)
  __builtin_amdgcn_s_wait_asynccnt(0);
#  else
  asm volatile("s_wait_asynccnt 0" ::: "memory");
#  endif
#endif
}

// ---------------------------------------------------------------------------
static __device__ __forceinline__ v8f wmma_f16(v16h a, v16h b, v8f c) {
  return __builtin_amdgcn_wmma_f32_16x16x32_f16(
      /*neg_a=*/false, a, /*neg_b=*/false, b,
      /*c_mod=*/(short)0, c, /*reuse_a=*/false, /*reuse_b=*/false);
}

// Load 32 bytes (16 halves) as two 16B chunks separated by `gap` halves.
static __device__ __forceinline__ v16h ld2x16B(const _Float16* p, int gap) {
  struct alignas(16) Raw { uint4 lo; uint4 hi; };
  Raw r;
  r.lo = *(const uint4*)p;
  r.hi = *(const uint4*)(p + gap);
  return __builtin_bit_cast(v16h, r);
}

// A-operand fragment (16x32 f16 tile, row-major source, stride 320), per the
// CDNA5 ISA 16-bit A-matrix VGPR layout: lane (h = lane>>4) holds halves
// K = {8h..8h+7, 16+8h..23+8h} of row m = row0 + (lane&15).
static __device__ __forceinline__ v16h load_a_frag(const _Float16* __restrict__ P,
                                                   int row0, int k0) {
  const int lane = threadIdx.x & 31;
  const int m = row0 + (lane & 15);
  const int h = lane >> 4;
  return ld2x16B(P + m * NYD + k0 + 8 * h, 16);
}

// B-operand fragment (32x16 f16 tile) from the n-major LDS stage:
// lane (h = lane>>4) holds K = {8h..8h+7, 16+8h..23+8h} of output column n = nbase+(lane&15)
// -> 2x ds_load_b128, conflict-free with LDS_ROW=40.
static __device__ __forceinline__ v16h load_b_frag_lds(const _Float16* s, int nbase) {
  const int lane = threadIdx.x & 31;
  const int n = nbase + (lane & 15);
  const int h = lane >> 4;
  return ld2x16B(s + n * LDS_ROW + 8 * h, 16);
}

// Stage a 32(K) x 32(N) B tile (real+imag) into LDS. B source is
// column-major: row n of Q is output column n (Xt for gemm1; A symmetric for
// gemm2). 128 threads x 16B per component.
static __device__ __forceinline__ void stage_b(const _Float16* __restrict__ Qr,
                                               const _Float16* __restrict__ Qi,
                                               _Float16* sBr, _Float16* sBi,
                                               int col0, int k0) {
  const int tid = threadIdx.x;   // 0..127
  const int n = tid >> 2;        // 0..31
  const int q = tid & 3;         // 0..3 (16B chunk within 64B row)
  const int gofs = (col0 + n) * NYD + k0 + q * 8;
  const int lofs = n * LDS_ROW + q * 8;
  copy16B_to_lds(Qr + gofs, sBr + lofs);
  copy16B_to_lds(Qi + gofs, sBi + lofs);
}

// ---------------------------------------------------------------------------
// Build centered orthonormal DFT matrix A (symmetric):
//   A[k][n] = exp(-2*pi*i * ((k+160)*(n+160) mod 320) / 320) / sqrt(320)
__global__ void batchelor_build_dft(_Float16* __restrict__ Ar,
                                    _Float16* __restrict__ Ai) {
  const int k = blockIdx.x;
  const int n = threadIdx.x;
  const int p = ((k + 160) * (n + 160)) % NXD;
  const float ang = -6.28318530717958647692f * (float)p / (float)NXD;
  const float scale = 0.05590169943749474241f; // 1/sqrt(320)
  Ar[k * NYD + n] = (_Float16)(cosf(ang) * scale);
  Ai[k * NYD + n] = (_Float16)(sinf(ang) * scale);
}

__global__ void batchelor_zero_out(float* __restrict__ out, int n) {
  const int i = blockIdx.x * blockDim.x + threadIdx.x;
  if (i < n) out[i] = 0.0f;
}

// ---------------------------------------------------------------------------
// Per-frame: bilinear warp of complex image, multiply by all 12 coil maps,
// emit f16 X planes COLUMN-MAJOR: Xt[c][y*320 + x]  (coalesced stores; makes
// the GEMM1 B-tile staging fully contiguous).
__global__ void batchelor_xprep(const float* __restrict__ img_r,
                                const float* __restrict__ img_i,
                                const float* __restrict__ smaps_r,
                                const float* __restrict__ smaps_i,
                                const float* __restrict__ flow,
                                int t,
                                _Float16* __restrict__ Xr_t,
                                _Float16* __restrict__ Xi_t) {
  const int idx = blockIdx.x * blockDim.x + threadIdx.x; // = y*320 + x
  if (idx >= NPIX) return;
  const int y = idx / NXD;
  const int x = idx % NXD;
  const int pix = x * NYD + y;          // row-major pixel index

  const float fx = flow[(pix * 2 + 0) * NTD + t];
  const float fy = flow[(pix * 2 + 1) * NTD + t];
  const float xx = fminf(fmaxf((float)x + fx, 0.0f), (float)(NXD - 1));
  const float yy = fminf(fmaxf((float)y + fy, 0.0f), (float)(NYD - 1));
  const float xf = floorf(xx), yf = floorf(yy);
  const float wx = xx - xf,   wy = yy - yf;
  const int x0 = (int)xf, y0 = (int)yf;
  const int x1 = min(x0 + 1, NXD - 1);
  const int y1 = min(y0 + 1, NYD - 1);
  const float w00 = (1.0f - wx) * (1.0f - wy);
  const float w10 = wx * (1.0f - wy);
  const float w01 = (1.0f - wx) * wy;
  const float w11 = wx * wy;
  const int i00 = x0 * NYD + y0, i10 = x1 * NYD + y0;
  const int i01 = x0 * NYD + y1, i11 = x1 * NYD + y1;

  const float wr = w00 * img_r[i00] + w10 * img_r[i10] + w01 * img_r[i01] + w11 * img_r[i11];
  const float wi = w00 * img_i[i00] + w10 * img_i[i10] + w01 * img_i[i01] + w11 * img_i[i11];

#pragma unroll
  for (int c = 0; c < NCD; ++c) {
    const float sr = smaps_r[pix * NCD + c];
    const float si = smaps_i[pix * NCD + c];
    Xr_t[c * NPIX + idx] = (_Float16)(wr * sr - wi * si);
    Xi_t[c * NPIX + idx] = (_Float16)(wr * si + wi * sr);
  }
}

// ---------------------------------------------------------------------------
// Complex GEMM core: 4 waves/block, wave owns M-tile (blockIdx.x*4+wave),
// and a 16x32 output strip (two N-tiles). B tile (32K x 32N, r+i) staged once
// per K-step in LDS and shared by all 4 waves. 8 WMMAs per wave per K-step.
// acc[0..3] = rr0,ii0,ri0,ir0 (cols col0..col0+15)
// acc[4..7] = rr1,ii1,ri1,ir1 (cols col0+16..col0+31)
static __device__ __forceinline__ void
cgemm_core(const _Float16* __restrict__ Pr, const _Float16* __restrict__ Pi,
           const _Float16* __restrict__ Qr, const _Float16* __restrict__ Qi,
           int row0, int col0, _Float16* sBr, _Float16* sBi, v8f acc[8]) {
#pragma unroll
  for (int i = 0; i < 8; ++i) acc[i] = (v8f){};

  for (int k0 = 0; k0 < NXD; k0 += 32) {
    __syncthreads();                       // WAR guard on LDS tiles
    stage_b(Qr, Qi, sBr, sBi, col0, k0);
    if (k0 + 32 < NXD) {                   // prefetch next A k-slice
      const int lane = threadIdx.x & 31;
      __builtin_prefetch(Pr + (row0 + (lane & 15)) * NYD + k0 + 32, 0, 1);
    }
    wait_async_lds();
    __syncthreads();

    const v16h ar  = load_a_frag(Pr, row0, k0);
    const v16h ai  = load_a_frag(Pi, row0, k0);
    const v16h br0 = load_b_frag_lds(sBr, 0);
    const v16h bi0 = load_b_frag_lds(sBi, 0);
    const v16h br1 = load_b_frag_lds(sBr, 16);
    const v16h bi1 = load_b_frag_lds(sBi, 16);

    acc[0] = wmma_f16(ar, br0, acc[0]);
    acc[1] = wmma_f16(ai, bi0, acc[1]);
    acc[2] = wmma_f16(ar, bi0, acc[2]);
    acc[3] = wmma_f16(ai, br0, acc[3]);
    acc[4] = wmma_f16(ar, br1, acc[4]);
    acc[5] = wmma_f16(ai, bi1, acc[5]);
    acc[6] = wmma_f16(ar, bi1, acc[6]);
    acc[7] = wmma_f16(ai, br1, acc[7]);
  }
}

// Stage 1: T_c = A * X_c   (f32 acc -> f16 out, row-major T)
__global__ __launch_bounds__(128)
void batchelor_gemm1(const _Float16* __restrict__ Ar, const _Float16* __restrict__ Ai,
                     const _Float16* __restrict__ Xr, const _Float16* __restrict__ Xi,
                     _Float16* __restrict__ Tr, _Float16* __restrict__ Ti) {
  __shared__ alignas(16) _Float16 sBr[32 * LDS_ROW];
  __shared__ alignas(16) _Float16 sBi[32 * LDS_ROW];
  const int wave = threadIdx.x >> 5;                 // 0..3
  const int row0 = (blockIdx.x * 4 + wave) * 16;     // 20 M-tiles exactly
  const int col0 = blockIdx.y * 32;                  // 10 N-pairs exactly
  const int c    = blockIdx.z;

  v8f acc[8];
  cgemm_core(Ar, Ai, Xr + c * NPIX, Xi + c * NPIX, row0, col0, sBr, sBi, acc);

  const int lane = threadIdx.x & 31;
  const int ncol = lane & 15;
  const int h    = lane >> 4;
  _Float16* tr = Tr + c * NPIX;
  _Float16* ti = Ti + c * NPIX;
#pragma unroll
  for (int jt = 0; jt < 2; ++jt) {
    const int n = col0 + jt * 16 + ncol;
#pragma unroll
    for (int r = 0; r < 8; ++r) {
      const int m = row0 + r + 8 * h;
      tr[m * NYD + n] = (_Float16)(acc[4 * jt + 0][r] - acc[4 * jt + 1][r]);
      ti[m * NYD + n] = (_Float16)(acc[4 * jt + 2][r] + acc[4 * jt + 3][r]);
    }
  }
}

// Stage 2: Y = T_c * A ; out_c += mask_t .* Y  (frames serialized -> no atomics)
__global__ __launch_bounds__(128)
void batchelor_gemm2(const _Float16* __restrict__ Tr, const _Float16* __restrict__ Ti,
                     const _Float16* __restrict__ Ar, const _Float16* __restrict__ Ai,
                     const float* __restrict__ mask, int t,
                     float* __restrict__ out) {
  __shared__ alignas(16) _Float16 sBr[32 * LDS_ROW];
  __shared__ alignas(16) _Float16 sBi[32 * LDS_ROW];
  const int wave = threadIdx.x >> 5;
  const int row0 = (blockIdx.x * 4 + wave) * 16;
  const int col0 = blockIdx.y * 32;
  const int c    = blockIdx.z;

  v8f acc[8];
  cgemm_core(Tr + c * NPIX, Ti + c * NPIX, Ar, Ai, row0, col0, sBr, sBi, acc);

  const int lane = threadIdx.x & 31;
  const int ncol = lane & 15;
  const int h    = lane >> 4;
#pragma unroll
  for (int jt = 0; jt < 2; ++jt) {
    const int n = col0 + jt * 16 + ncol;
#pragma unroll
    for (int r = 0; r < 8; ++r) {
      const int m   = row0 + r + 8 * h;
      const int pix = m * NYD + n;
      const float msk = mask[pix * NTD + t];
      const float yr = acc[4 * jt + 0][r] - acc[4 * jt + 1][r];
      const float yi = acc[4 * jt + 2][r] + acc[4 * jt + 3][r];
      float* o = out + (pix * NCD + c) * 2;
      o[0] += msk * yr;
      o[1] += msk * yi;
    }
  }
}

// ---------------------------------------------------------------------------
extern "C" void kernel_launch(void* const* d_in, const int* in_sizes, int n_in,
                              void* d_out, int out_size, void* d_ws, size_t ws_size,
                              hipStream_t stream) {
  (void)in_sizes; (void)n_in; (void)out_size; (void)ws_size;
  const float* image_r = (const float*)d_in[0];
  const float* image_i = (const float*)d_in[1];
  const float* mask    = (const float*)d_in[2];
  const float* smaps_r = (const float*)d_in[3];
  const float* smaps_i = (const float*)d_in[4];
  const float* flow    = (const float*)d_in[5];
  float* out = (float*)d_out;

  char* ws = (char*)d_ws;
  _Float16* Ar = (_Float16*)(ws + 0);          //   204,800 B
  _Float16* Ai = (_Float16*)(ws + 204800);     //   204,800 B
  _Float16* Xr = (_Float16*)(ws + 409600);     // 2,457,600 B (column-major per coil)
  _Float16* Xi = (_Float16*)(ws + 2867200);    // 2,457,600 B
  _Float16* Tr = (_Float16*)(ws + 5324800);    // 2,457,600 B (row-major per coil)
  _Float16* Ti = (_Float16*)(ws + 7782400);    // 2,457,600 B  (total ~10.24 MB)

  batchelor_build_dft<<<dim3(NXD), dim3(NYD), 0, stream>>>(Ar, Ai);
  const int outElems = NPIX * NCD * 2;
  batchelor_zero_out<<<dim3((outElems + 255) / 256), dim3(256), 0, stream>>>(out, outElems);

  const dim3 gemmGrid(5, 10, NCD);   // 5*4 waves = 20 M-tiles, 10 N-pairs, 12 coils
  const dim3 gemmBlock(128);

  for (int t = 0; t < NTD; ++t) {
    batchelor_xprep<<<dim3((NPIX + 255) / 256), dim3(256), 0, stream>>>(
        image_r, image_i, smaps_r, smaps_i, flow, t, Xr, Xi);
    batchelor_gemm1<<<gemmGrid, gemmBlock, 0, stream>>>(Ar, Ai, Xr, Xi, Tr, Ti);
    batchelor_gemm2<<<gemmGrid, gemmBlock, 0, stream>>>(Tr, Ti, Ar, Ai, mask, t, out);
  }
}
